// Qwen3_5GatedDeltaNet_85529978732977
// MI455X (gfx1250) — matrix operand order repaired and checked by reading, not compiled, not run on hardware
//
#include <hip/hip_runtime.h>
#include <hip/hip_bf16.h>
#include <math.h>

typedef __attribute__((ext_vector_type(16))) __bf16 v16bf;
typedef __attribute__((ext_vector_type(8)))  float  v8f;
typedef __attribute__((ext_vector_type(4)))  unsigned int u32x4_t;
typedef __attribute__((ext_vector_type(8)))  int i32x8_t;
typedef __attribute__((ext_vector_type(4)))  int i32x4_t;

#define S_LEN 512
#define H_DIM 2048
#define NVH 16
#define NKH 8
#define DK 128
#define DV 128
#define KW 4
#define KEYD 1024
#define VALD 2048
#define CONVD 4096

// ---------------------------------------------------------------------------
// 0) hs = hidden * mask
// ---------------------------------------------------------------------------
__global__ void mask_mul_kernel(const float* __restrict__ hidden,
                                const int* __restrict__ mask,
                                float* __restrict__ hs) {
    int i = blockIdx.x * 256 + threadIdx.x;
    if (i < S_LEN * H_DIM) {
        int s = i / H_DIM;
        hs[i] = hidden[i] * (float)mask[s];
    }
}

// ---------------------------------------------------------------------------
// TDM: async-load a 2D f32 tile [tile_h x tile_w] (row stride row_stride
// elements in global) into LDS at lds_off, with hardware padding of
// pad_amount(code) dwords every pad_interval(code) dwords (ISA 8.3/8.4).
// D# group0: count=1 | lds_addr | global_addr(57b) | type=2.
// D# group1: data_size=4B, pad_enable, tensor dims, tile dims, dim0 stride.
// Groups 2/3 (and trailing group) zero: tile is 2-D.
// ---------------------------------------------------------------------------
__device__ __forceinline__ void tdm_load_tile_f32(
    unsigned lds_off, const float* gptr, int tile_w, int tile_h,
    int row_stride, int tdim0, int tdim1, int pad_int_code, int pad_amt_code) {
    unsigned long long ga = (unsigned long long)(uintptr_t)gptr;
    u32x4_t g0;
    g0[0] = 1u;                                   // count = 1 (valid D#)
    g0[1] = lds_off;                              // lds_addr (bytes)
    g0[2] = (unsigned)ga;                         // global_addr[31:0]
    g0[3] = (unsigned)((ga >> 32) & 0x1FFFFFFu)   // global_addr[56:32]
          | (2u << 30);                           // type = 2 (image)
    i32x8_t g1;
    g1[0] = (2 << 16)                             // data_size = 4 bytes
          | (1 << 20)                             // pad_enable
          | (pad_int_code << 22)                  // pad interval (2^(c+1) dw)
          | (pad_amt_code << 25);                 // pad amount (c+1 dwords)
    g1[1] = (int)(((unsigned)tdim0 & 0xFFFFu) << 16);            // tensor_dim0 lo
    g1[2] = (int)((((unsigned)tdim0 >> 16) & 0xFFFFu)
          | (((unsigned)tdim1 & 0xFFFFu) << 16));                // dim0 hi|dim1 lo
    g1[3] = (int)((((unsigned)tdim1 >> 16) & 0xFFFFu)
          | ((unsigned)tile_w << 16));                           // dim1 hi|tile_dim0
    g1[4] = tile_h & 0xFFFF;                                     // tile_dim1
    g1[5] = row_stride;                                          // dim0_stride[31:0]
    g1[6] = 0;
    g1[7] = 0;
    i32x4_t z4 = {0, 0, 0, 0};
    i32x8_t z8 = {0, 0, 0, 0, 0, 0, 0, 0};
    __builtin_amdgcn_tensor_load_to_lds(g0, g1, z4, z4, z8, 0);
}

// ---------------------------------------------------------------------------
// 1) WMMA GEMM, fp32 in/out via bf16x3 split.  C[M,N] = A[M,K] * B[K,N].
//    WG tile 64(M) x 128(N), 256 threads = 8 waves, wave = 16x64 strip.
//    Tiles staged by the Tensor Data Mover, double-buffered in LDS.
// ---------------------------------------------------------------------------
__global__ __launch_bounds__(256) void wmma_gemm_bf16x3(
    const float* __restrict__ A, const float* __restrict__ B,
    float* __restrict__ C, int M, int N, int Kd) {
    __shared__ __align__(16) float As[2][64 * 36];    // 32 cols, stride 36
    __shared__ __align__(16) float Bs[2][32 * 132];   // 128 cols, stride 132

    const int tid  = threadIdx.x;
    const int wave = tid >> 5;
    const int lane = tid & 31;
    const int wm   = wave >> 1;      // 0..3
    const int wn   = wave & 1;       // 0..1
    const int lh   = lane >> 4;      // lane half (0/1)
    const int l15  = lane & 15;
    const int tileM = blockIdx.y * 64;
    const int tileN = blockIdx.x * 128;
    const bool issuer = (tid < 32);  // wave 0 drives the TDM

    v8f acc[4];
#pragma unroll
    for (int n = 0; n < 4; ++n) acc[n] = (v8f){0,0,0,0,0,0,0,0};

    const unsigned offA0 = (unsigned)(uintptr_t)&As[0][0];
    const unsigned offA1 = (unsigned)(uintptr_t)&As[1][0];
    const unsigned offB0 = (unsigned)(uintptr_t)&Bs[0][0];
    const unsigned offB1 = (unsigned)(uintptr_t)&Bs[1][0];

    // prologue: async-load k-block 0 into buffer 0
    if (issuer) {
        tdm_load_tile_f32(offA0, &A[(size_t)tileM * Kd], 32, 64, Kd,
                          Kd, M, /*interval 32dw*/4, /*pad 4dw*/3);
        tdm_load_tile_f32(offB0, &B[tileN], 128, 32, N,
                          N, Kd, /*interval 128dw*/6, /*pad 4dw*/3);
    }

    const int nk = Kd >> 5;
    for (int kb = 0; kb < nk; ++kb) {
        const int cur = kb & 1;
        if (issuer) {
            if (kb + 1 < nk) {   // issue next tile, then wait only for current
                const int k1 = (kb + 1) << 5;
                tdm_load_tile_f32(cur ? offA0 : offA1,
                                  &A[(size_t)tileM * Kd + k1], 32, 64, Kd,
                                  Kd, M, 4, 3);
                tdm_load_tile_f32(cur ? offB0 : offB1,
                                  &B[(size_t)k1 * N + tileN], 128, 32, N,
                                  N, Kd, 6, 3);
                __builtin_amdgcn_s_wait_tensorcnt(2);
            } else {
                __builtin_amdgcn_s_wait_tensorcnt(0);
            }
        }
        __syncthreads();

        // ---- build A fragment (16x32 bf16, ISA 7.12.2 layout), hi/lo split
        const float* Ab = &As[cur][0];
        const float* Bb = &Bs[cur][0];
        v16bf ah, al;
#pragma unroll
        for (int e = 0; e < 16; ++e) {
            int p  = e >> 1;
            int kk = ((p >> 2) << 4) + (lh << 3) + ((p & 3) << 1) + (e & 1);
            float f = Ab[(wm * 16 + l15) * 36 + kk];
            __bf16 h = (__bf16)f;
            ah[e] = h;
            al[e] = (__bf16)(f - (float)h);
        }
        // ---- 4 N-subtiles of 16: load B frag, 3 WMMAs each (bf16x3)
#pragma unroll
        for (int n = 0; n < 4; ++n) {
            v16bf bh, bl;
#pragma unroll
            for (int e = 0; e < 16; ++e) {
                int kk = ((e >> 3) << 4) + (lh << 3) + (e & 7);   // B: lanes 0-15 K=0..7,16..23; 16-31 K=8..15,24..31
                float f = Bb[kk * 132 + wn * 64 + n * 16 + l15];
                __bf16 h = (__bf16)f;
                bh[e] = h;
                bl[e] = (__bf16)(f - (float)h);
            }
            v8f a = acc[n];
            a = __builtin_amdgcn_wmma_f32_16x16x32_bf16(false, ah, false, bh,
                                                        (short)0, a, false, false);
            a = __builtin_amdgcn_wmma_f32_16x16x32_bf16(false, ah, false, bl,
                                                        (short)0, a, false, false);
            a = __builtin_amdgcn_wmma_f32_16x16x32_bf16(false, al, false, bh,
                                                        (short)0, a, false, false);
            acc[n] = a;
        }
        __syncthreads();   // protects buffer reuse two iterations out
    }

    // ---- writeback: C/D layout => row M = lh*8 + v, col N = l15
#pragma unroll
    for (int n = 0; n < 4; ++n) {
#pragma unroll
        for (int v = 0; v < 8; ++v) {
            int gm = tileM + wm * 16 + lh * 8 + v;
            int gn = tileN + wn * 64 + n * 16 + l15;
            C[(size_t)gm * N + gn] = acc[n][v];
        }
    }
}

// ---------------------------------------------------------------------------
// 2) small projections W_b/W_a (N=16 each) fused with beta/g computation
// ---------------------------------------------------------------------------
__global__ __launch_bounds__(128) void small_proj_kernel(
    const float* __restrict__ hs, const float* __restrict__ Wb,
    const float* __restrict__ Wa, const float* __restrict__ A_log,
    const float* __restrict__ dt_bias, float* __restrict__ beta,
    float* __restrict__ g) {
    __shared__ __align__(16) float row[H_DIM];
    __shared__ float partial[32];
    int s = blockIdx.x;
    int t = threadIdx.x;
    for (int i = t; i < H_DIM / 4; i += 128)
        reinterpret_cast<float4*>(row)[i] =
            reinterpret_cast<const float4*>(hs + (size_t)s * H_DIM)[i];
    if (t < 32) partial[t] = 0.f;
    __syncthreads();

    int j = t & 31, part = t >> 5;
    const float* W = (j < 16) ? Wb : Wa;
    int jj = j & 15;
    float sum = 0.f;
    for (int k = part * 512; k < part * 512 + 512; ++k)
        sum += row[k] * W[(size_t)k * 16 + jj];
    atomicAdd(&partial[j], sum);          // ds_add_f32
    __syncthreads();

    if (t < 16) {
        beta[s * NVH + t] = 1.f / (1.f + __expf(-partial[t]));
    } else if (t < 32) {
        int h = t - 16;
        float x  = partial[t] + dt_bias[h];
        float sp = (x > 20.f) ? x : log1pf(__expf(x));
        g[s * NVH + h] = -__expf(A_log[h]) * sp;
    }
}

// ---------------------------------------------------------------------------
// 3) depthwise causal conv1d (K=4) + SiLU
// ---------------------------------------------------------------------------
__global__ void conv_silu_kernel(const float* __restrict__ pre,
                                 const float* __restrict__ w,   // [4,1,4096]
                                 float* __restrict__ out) {
    int i = blockIdx.x * 256 + threadIdx.x;
    if (i >= S_LEN * CONVD) return;
    int c = i & (CONVD - 1);
    int s = i >> 12;
    float acc = 0.f;
#pragma unroll
    for (int t = 0; t < KW; ++t) {
        int ss = s + t - (KW - 1);
        if (ss >= 0) acc += w[t * CONVD + c] * pre[(size_t)ss * CONVD + c];
    }
    out[i] = acc / (1.f + __expf(-acc));   // x*sigmoid(x)
}

// ---------------------------------------------------------------------------
// 4) l2-norm of q,k (per (s, key-head)); q scaled by 1/sqrt(DK)
// ---------------------------------------------------------------------------
__global__ __launch_bounds__(128) void qk_norm_kernel(
    const float* __restrict__ mixed, float* __restrict__ qn,
    float* __restrict__ kn) {
    int wid  = blockIdx.x * 4 + (threadIdx.x >> 5);
    int lane = threadIdx.x & 31;
    int s = wid >> 3, kh = wid & 7;
    const float* qr = mixed + (size_t)s * CONVD + kh * DK;
    const float* kr = mixed + (size_t)s * CONVD + KEYD + kh * DK;
    float qv[4], kv[4], sq = 0.f, sk = 0.f;
#pragma unroll
    for (int i = 0; i < 4; ++i) {
        qv[i] = qr[lane + 32 * i]; kv[i] = kr[lane + 32 * i];
        sq += qv[i] * qv[i];       sk += kv[i] * kv[i];
    }
#pragma unroll
    for (int m = 16; m >= 1; m >>= 1) {
        sq += __shfl_xor(sq, m, 32);
        sk += __shfl_xor(sk, m, 32);
    }
    float rq = rsqrtf(sq + 1e-6f) * 0.08838834764831845f;  // * 1/sqrt(128)
    float rk = rsqrtf(sk + 1e-6f);
    float* qo = qn + (size_t)s * NKH * DK + kh * DK;
    float* ko = kn + (size_t)s * NKH * DK + kh * DK;
#pragma unroll
    for (int i = 0; i < 4; ++i) {
        qo[lane + 32 * i] = qv[i] * rq;
        ko[lane + 32 * i] = kv[i] * rk;
    }
}

// ---------------------------------------------------------------------------
// 5) gated delta-rule scan. grid = 64 WGs: (head, dv-quarter). 128 threads:
//    thread = (dv_local = t>>2, dk quarter = t&3), 32 fp32 state regs each.
// ---------------------------------------------------------------------------
__global__ __launch_bounds__(128) void delta_scan_kernel(
    const float* __restrict__ qn, const float* __restrict__ kn,
    const float* __restrict__ mixed, const float* __restrict__ g,
    const float* __restrict__ beta, float* __restrict__ core) {
    __shared__ __align__(16) float ks[DK];
    __shared__ __align__(16) float qs[DK];
    __shared__ float vs[32];
    const int h   = blockIdx.x >> 2;
    const int dvb = (blockIdx.x & 3) * 32;
    const int t   = threadIdx.x;
    const int kh  = h >> 1;                 // repeat(2) of key heads
    const int dvl = t >> 2, quarter = t & 3;

    float st[32];
#pragma unroll
    for (int i = 0; i < 32; ++i) st[i] = 0.f;

    const float* kb = kn + (size_t)kh * DK;
    const float* qb = qn + (size_t)kh * DK;
    const float* vb = mixed + 2 * KEYD + h * DV + dvb;

    for (int s = 0; s < S_LEN; ++s) {
        ks[t] = kb[(size_t)s * NKH * DK + t];
        qs[t] = qb[(size_t)s * NKH * DK + t];
        if (t < 32) vs[t] = vb[(size_t)s * CONVD + t];
        __syncthreads();

        float eg = __expf(g[s * NVH + h]);
        float bt = beta[s * NVH + h];

        const float4* k4 = reinterpret_cast<const float4*>(&ks[quarter * 32]);
        const float4* q4 = reinterpret_cast<const float4*>(&qs[quarter * 32]);
        float kk[32];
#pragma unroll
        for (int i8 = 0; i8 < 8; ++i8) {
            float4 kf = k4[i8];
            kk[i8*4+0] = kf.x; kk[i8*4+1] = kf.y;
            kk[i8*4+2] = kf.z; kk[i8*4+3] = kf.w;
        }
        float kvsum = 0.f;
#pragma unroll
        for (int i = 0; i < 32; ++i) { st[i] *= eg; kvsum += st[i] * kk[i]; }
        kvsum += __shfl_xor(kvsum, 1, 32);
        kvsum += __shfl_xor(kvsum, 2, 32);

        float delta = (vs[dvl] - kvsum) * bt;

        float osum = 0.f;
#pragma unroll
        for (int i8 = 0; i8 < 8; ++i8) {
            float4 qf = q4[i8];
            float qq[4] = {qf.x, qf.y, qf.z, qf.w};
#pragma unroll
            for (int j = 0; j < 4; ++j) {
                int i = i8 * 4 + j;
                st[i] += kk[i] * delta;
                osum  += st[i] * qq[j];
            }
        }
        osum += __shfl_xor(osum, 1, 32);
        osum += __shfl_xor(osum, 2, 32);
        if (quarter == 0)
            core[(size_t)s * VALD + h * DV + dvb + dvl] = osum;
        __syncthreads();
    }
}

// ---------------------------------------------------------------------------
// 6) gated RMSNorm (over DV) * norm_weight * silu(z)
// ---------------------------------------------------------------------------
__global__ __launch_bounds__(128) void rms_gate_kernel(
    const float* __restrict__ core, const float* __restrict__ z,
    const float* __restrict__ nw, float* __restrict__ out) {
    int wid  = blockIdx.x * 4 + (threadIdx.x >> 5);
    int lane = threadIdx.x & 31;
    int s = wid >> 4, h = wid & 15;
    const float* row  = core + (size_t)s * VALD + h * DV;
    const float* zrow = z    + (size_t)s * VALD + h * DV;
    float v[4], ss = 0.f;
#pragma unroll
    for (int i = 0; i < 4; ++i) { v[i] = row[lane + 32 * i]; ss += v[i] * v[i]; }
#pragma unroll
    for (int m = 16; m >= 1; m >>= 1) ss += __shfl_xor(ss, m, 32);
    float r = rsqrtf(ss * (1.f / DV) + 1e-6f);
    float* o = out + (size_t)s * VALD + h * DV;
#pragma unroll
    for (int i = 0; i < 4; ++i) {
        int idx = lane + 32 * i;
        float zz = zrow[idx];
        o[idx] = v[i] * r * nw[idx] * (zz / (1.f + __expf(-zz)));
    }
}

// ---------------------------------------------------------------------------
extern "C" void kernel_launch(void* const* d_in, const int* in_sizes, int n_in,
                              void* d_out, int out_size, void* d_ws, size_t ws_size,
                              hipStream_t stream) {
    const float* hidden = (const float*)d_in[0];
    const float* W_qkv  = (const float*)d_in[1];
    const float* W_z    = (const float*)d_in[2];
    const float* W_b    = (const float*)d_in[3];
    const float* W_a    = (const float*)d_in[4];
    const float* convw  = (const float*)d_in[5];
    const float* dtb    = (const float*)d_in[6];
    const float* A_log  = (const float*)d_in[7];
    const float* nw     = (const float*)d_in[8];
    const float* W_out  = (const float*)d_in[9];
    const int*   mask   = (const int*)d_in[10];

    char* ws = (char*)d_ws;
    const size_t MB = 1024 * 1024;
    float* hs        = (float*)(ws + 0);          // 4 MB
    float* mixed_pre = (float*)(ws + 4  * MB);    // 8 MB (reused as core_g)
    float* mixed_c   = (float*)(ws + 12 * MB);    // 8 MB
    float* z_buf     = (float*)(ws + 20 * MB);    // 4 MB
    float* qn        = (float*)(ws + 24 * MB);    // 2 MB
    float* kn        = (float*)(ws + 26 * MB);    // 2 MB
    float* beta      = (float*)(ws + 28 * MB);    // 32 KB
    float* g         = (float*)(ws + 28 * MB + 64 * 1024);
    float* core      = (float*)(ws + 29 * MB);    // 4 MB
    float* core_g    = mixed_pre;                 // reuse (free after conv)
    float* out       = (float*)d_out;

    // 0) mask
    mask_mul_kernel<<<(S_LEN * H_DIM) / 256, 256, 0, stream>>>(hidden, mask, hs);
    // 1) QKV projection: [512,2048]x[2048,4096]
    wmma_gemm_bf16x3<<<dim3(CONVD / 128, S_LEN / 64), 256, 0, stream>>>(
        hs, W_qkv, mixed_pre, S_LEN, CONVD, H_DIM);
    // 2) z projection: [512,2048]x[2048,2048]
    wmma_gemm_bf16x3<<<dim3(VALD / 128, S_LEN / 64), 256, 0, stream>>>(
        hs, W_z, z_buf, S_LEN, VALD, H_DIM);
    // 3) b/a projections + beta/g
    small_proj_kernel<<<S_LEN, 128, 0, stream>>>(hs, W_b, W_a, A_log, dtb, beta, g);
    // 4) conv + silu
    conv_silu_kernel<<<(S_LEN * CONVD) / 256, 256, 0, stream>>>(mixed_pre, convw, mixed_c);
    // 5) q/k l2norm
    qk_norm_kernel<<<(S_LEN * NKH) / 4, 128, 0, stream>>>(mixed_c, qn, kn);
    // 6) delta-rule scan (16 heads x 4 dv-quarters)
    delta_scan_kernel<<<NVH * 4, 128, 0, stream>>>(qn, kn, mixed_c, g, beta, core);
    // 7) gated rmsnorm
    rms_gate_kernel<<<(S_LEN * NVH) / 4, 128, 0, stream>>>(core, z_buf, nw, core_g);
    // 8) output projection: [512,2048]x[2048,2048]
    wmma_gemm_bf16x3<<<dim3(H_DIM / 128, S_LEN / 64), 256, 0, stream>>>(
        core_g, W_out, out, S_LEN, H_DIM, VALD);
}
